// TransformerLayerMoE_58720792871053
// MI455X (gfx1250) — matrix operand order repaired and checked by reading, not compiled, not run on hardware
//
#include <hip/hip_runtime.h>
#include <cmath>

// ---------------- CDNA5 WMMA types / helpers ----------------
typedef _Float16 h16 __attribute__((ext_vector_type(16)));
typedef _Float16 h8  __attribute__((ext_vector_type(8)));
typedef float    f8  __attribute__((ext_vector_type(8)));

#define DEVFN static __device__ __forceinline__

// A-fragment (f16, 16x32, row-major A with row stride = K):
// lane L: m = m0 + (L&15); elements [0..15] = K k0 + (L>>4)*16 + i
//         (contiguous order, matches B fragment)
// => two contiguous 16-byte runs per lane.
DEVFN h16 load_a32(const _Float16* p /* pre-offset: row + (lane>>4)*16 + k0 */) {
  h8 lo = *(const h8*)(p);
  h8 hi = *(const h8*)(p + 8);
  h16 r;
#pragma unroll
  for (int i = 0; i < 8; ++i) { r[i] = lo[i]; r[8 + i] = hi[i]; }
  return r;
}

// B-fragment (f16, 32x16 = KxN), stored as Bt[N][K] row-major (NT GEMM):
// lane L: n = n0 + (L&15); K = k0 + (L>>4)*16 + [0..15] -> one 32B load.
DEVFN f8 wmma16(h16 a, h16 b, f8 c) {
  return __builtin_amdgcn_wmma_f32_16x16x32_f16(
      /*neg_a=*/false, a, /*neg_b=*/false, b,
      /*c_mod=*/(short)0, c, /*reuse_a=*/false, /*reuse_b=*/false);
}

DEVFN f8 f8zero() { f8 z = {0.f, 0.f, 0.f, 0.f, 0.f, 0.f, 0.f, 0.f}; return z; }

// ---------------- problem constants ----------------
static constexpr int B = 4, S = 1024, D = 1024, F = 4096;
static constexpr int E = 8, H = 16, HD = 64, CAP = 160;
static constexpr int TOK = B * S;           // 4096 tokens
static constexpr int SLOTS = B * E * CAP;   // 5120 expert slots

// ---------------- conversion kernels ----------------
__global__ void cvt_f32_to_f16(const float* __restrict__ src,
                               _Float16* __restrict__ dst, long long n) {
  long long i = (long long)blockIdx.x * blockDim.x + threadIdx.x;
  long long stride = (long long)gridDim.x * blockDim.x;
  for (; i < n; i += stride) dst[i] = (_Float16)src[i];
}

// wi [E,D,F] -> wih [E,F,D]  (so GEMM1 is NT with K=D contiguous)
__global__ __launch_bounds__(256) void trans_wi_kernel(const float* __restrict__ wi,
                                                       _Float16* __restrict__ wih) {
  int id = blockIdx.x;            // e*D + d
  int e = id >> 10, d = id & 1023;
  const float* src = wi + ((long long)e * D + d) * F;
  for (int f = threadIdx.x; f < F; f += 256)
    wih[((long long)e * F + f) * D + d] = (_Float16)src[f];
}

// wo [E,F,D] -> woh [E,D,F]  (so GEMM2 is NT with K=F contiguous)
__global__ __launch_bounds__(256) void trans_wo_kernel(const float* __restrict__ wo,
                                                       _Float16* __restrict__ woh) {
  int id = blockIdx.x;            // e*F + f
  int e = id >> 12, f = id & 4095;
  const float* src = wo + ((long long)e * F + f) * D;
  for (int d = threadIdx.x; d < D; d += 256)
    woh[((long long)e * D + d) * F + f] = (_Float16)src[d];
}

// ---------------- generic NT GEMM: 32x64 tile per wave32 ----------------
// 2 A-fragments x 4 B-fragments -> 8 WMMA per k-step (1.5 b128 loads / WMMA)
// C[z][m][n] = act( sum_k A[z][m][k] * Bt[z&eMask][n][k] + bias[n] )
template <int ACT, int OUT16>
__global__ __launch_bounds__(32) void gemm_nt(
    const _Float16* __restrict__ A, const _Float16* __restrict__ Bt,
    const float* __restrict__ bias, void* __restrict__ Cout,
    int N, int K, long long aStride, long long bStride, long long cStride,
    int eMask) {
  const int z = blockIdx.z;
  A  += (long long)z * aStride;
  Bt += (long long)(z & eMask) * bStride;
  const int lane = threadIdx.x, l16 = lane & 15, hf = lane >> 4;
  const int m0 = blockIdx.x * 32, n0 = blockIdx.y * 64;
  const _Float16* ap0 = A + (long long)(m0 + l16) * K + hf * 16;
  const _Float16* ap1 = ap0 + (long long)16 * K;
  const _Float16* bp  = Bt + (long long)(n0 + l16) * K + hf * 16;
  const long long bstep = (long long)16 * K;
  f8 acc[2][4];
#pragma unroll
  for (int i = 0; i < 2; ++i)
#pragma unroll
    for (int j = 0; j < 4; ++j) acc[i][j] = f8zero();
  for (int k0 = 0; k0 < K; k0 += 32) {
    __builtin_prefetch(bp + k0 + 512, 0, 3);           // WGP-scope prefetch
    __builtin_prefetch(bp + 2 * bstep + k0 + 512, 0, 3);
    h16 a0 = load_a32(ap0 + k0);
    h16 a1 = load_a32(ap1 + k0);
    h16 b0 = *(const h16*)(bp + k0);
    h16 b1 = *(const h16*)(bp + bstep + k0);
    h16 b2 = *(const h16*)(bp + 2 * bstep + k0);
    h16 b3 = *(const h16*)(bp + 3 * bstep + k0);
    acc[0][0] = wmma16(a0, b0, acc[0][0]);
    acc[0][1] = wmma16(a0, b1, acc[0][1]);
    acc[0][2] = wmma16(a0, b2, acc[0][2]);
    acc[0][3] = wmma16(a0, b3, acc[0][3]);
    acc[1][0] = wmma16(a1, b0, acc[1][0]);
    acc[1][1] = wmma16(a1, b1, acc[1][1]);
    acc[1][2] = wmma16(a1, b2, acc[1][2]);
    acc[1][3] = wmma16(a1, b3, acc[1][3]);
  }
#pragma unroll
  for (int j = 0; j < 4; ++j) {
    const int n = n0 + j * 16 + l16;
    const float bv = bias ? bias[n] : 0.0f;
#pragma unroll
    for (int i = 0; i < 2; ++i) {
#pragma unroll
      for (int r = 0; r < 8; ++r) {
        int m = m0 + i * 16 + r + hf * 8;
        float v = acc[i][j][r] + bv;
        if (ACT == 1) v = 0.5f * v * (1.0f + erff(v * 0.70710678118654752f));
        long long idx = (long long)z * cStride + (long long)m * N + n;
        if (OUT16) ((_Float16*)Cout)[idx] = (_Float16)v;
        else       ((float*)Cout)[idx] = v;
      }
    }
  }
}

// ---------------- QKV GEMM (32x64 tile) with scatter to attention layouts ----
// xh[4096,1024] @ winh[3072,1024]^T + b -> q/k [B,H,S,64] f16, vT [B,H,64,S] f16
__global__ __launch_bounds__(32) void qkv_kernel(
    const _Float16* __restrict__ xh, const _Float16* __restrict__ winh,
    const float* __restrict__ in_b, _Float16* __restrict__ qh,
    _Float16* __restrict__ kh, _Float16* __restrict__ vth) {
  const int lane = threadIdx.x, l16 = lane & 15, hf = lane >> 4;
  const int m0 = blockIdx.x * 32, n0 = blockIdx.y * 64;
  const _Float16* ap0 = xh + (long long)(m0 + l16) * D + hf * 16;
  const _Float16* ap1 = ap0 + (long long)16 * D;
  const _Float16* bp  = winh + (long long)(n0 + l16) * D + hf * 16;
  const long long bstep = (long long)16 * D;
  f8 acc[2][4];
#pragma unroll
  for (int i = 0; i < 2; ++i)
#pragma unroll
    for (int j = 0; j < 4; ++j) acc[i][j] = f8zero();
  for (int k0 = 0; k0 < D; k0 += 32) {
    __builtin_prefetch(bp + k0 + 512, 0, 3);
    __builtin_prefetch(bp + 2 * bstep + k0 + 512, 0, 3);
    h16 a0 = load_a32(ap0 + k0);
    h16 a1 = load_a32(ap1 + k0);
    h16 b0 = *(const h16*)(bp + k0);
    h16 b1 = *(const h16*)(bp + bstep + k0);
    h16 b2 = *(const h16*)(bp + 2 * bstep + k0);
    h16 b3 = *(const h16*)(bp + 3 * bstep + k0);
    acc[0][0] = wmma16(a0, b0, acc[0][0]);
    acc[0][1] = wmma16(a0, b1, acc[0][1]);
    acc[0][2] = wmma16(a0, b2, acc[0][2]);
    acc[0][3] = wmma16(a0, b3, acc[0][3]);
    acc[1][0] = wmma16(a1, b0, acc[1][0]);
    acc[1][1] = wmma16(a1, b1, acc[1][1]);
    acc[1][2] = wmma16(a1, b2, acc[1][2]);
    acc[1][3] = wmma16(a1, b3, acc[1][3]);
  }
#pragma unroll
  for (int j = 0; j < 4; ++j) {
    const int n = n0 + j * 16 + l16;
    const float bv = in_b[n];
    const int part = n >> 10, nn = n & 1023, head = nn >> 6, hd = nn & 63;
#pragma unroll
    for (int i = 0; i < 2; ++i) {
#pragma unroll
      for (int r = 0; r < 8; ++r) {
        int m = m0 + i * 16 + r + hf * 8, b = m >> 10, s = m & 1023;
        float v = acc[i][j][r] + bv;
        long long bh = (long long)(b * H + head);
        if (part == 0)      qh[(bh * S + s) * HD + hd] = (_Float16)(v * 0.125f);
        else if (part == 1) kh[(bh * S + s) * HD + hd] = (_Float16)v;
        else                vth[(bh * HD + hd) * S + s] = (_Float16)v;
      }
    }
  }
}

// ---------------- attention: one wave = 16 queries of one (b,h) ----------------
__global__ __launch_bounds__(32) void attn_kernel(
    const _Float16* __restrict__ qh, const _Float16* __restrict__ kh,
    const _Float16* __restrict__ vth, _Float16* __restrict__ ctxh) {
  __shared__ float sc[16 * 1024];   // 64 KB score staging (WGP has 320 KB)
  __shared__ float rowred[16];
  const int bh = blockIdx.y;        // b*H + h
  const int m0 = blockIdx.x * 16;   // query tile base
  const int lane = threadIdx.x, l16 = lane & 15, hf = lane >> 4;
  // Q A-fragments (16x64 split into two 16x32)
  const _Float16* qbase = qh + ((long long)bh * S + m0 + l16) * HD + hf * 16;
  h16 qa0 = load_a32(qbase);
  h16 qa1 = load_a32(qbase + 32);
  float cmax[8];
#pragma unroll
  for (int r = 0; r < 8; ++r) cmax[r] = -1e30f;
  // pass 1: scores -> LDS, track running row max
  for (int kb = 0; kb < S / 16; ++kb) {
    const _Float16* krow = kh + ((long long)bh * S + kb * 16 + l16) * HD;
    h16 b0 = *(const h16*)(krow + hf * 16);        // dims 0..31
    h16 b1 = *(const h16*)(krow + 32 + hf * 16);   // dims 32..63
    f8 st = f8zero();
    st = wmma16(qa0, b0, st);
    st = wmma16(qa1, b1, st);
#pragma unroll
    for (int r = 0; r < 8; ++r) {
      sc[(r + hf * 8) * 1024 + kb * 16 + l16] = st[r];
      cmax[r] = fmaxf(cmax[r], st[r]);
    }
  }
#pragma unroll
  for (int off = 1; off < 16; off <<= 1)
#pragma unroll
    for (int r = 0; r < 8; ++r)
      cmax[r] = fmaxf(cmax[r], __shfl_xor(cmax[r], off, 32));
  if (l16 == 0)
#pragma unroll
    for (int r = 0; r < 8; ++r) rowred[r + hf * 8] = cmax[r];
  __syncthreads();
  const float rm = rowred[l16];     // max for row m = l16
  // pass 2: P = exp(s - m) as A-fragment, O += P @ V^T
  f8 o0 = f8zero(), o1 = f8zero(), o2 = f8zero(), o3 = f8zero();
  float sumexp = 0.f;
  for (int kb = 0; kb < S / 32; ++kb) {
    const int kbase = kb * 32 + hf * 16;
    h16 pa;
#pragma unroll
    for (int i = 0; i < 8; ++i) {
      float p0 = __expf(sc[l16 * 1024 + kbase + i] - rm);
      float p1 = __expf(sc[l16 * 1024 + kbase + 8 + i] - rm);
      sumexp += p0 + p1;
      pa[i] = (_Float16)p0;
      pa[8 + i] = (_Float16)p1;
    }
    const _Float16* vbase = vth + (long long)bh * HD * S + kb * 32 + hf * 16;
    o0 = wmma16(pa, *(const h16*)(vbase + (long long)(l16)      * S), o0);
    o1 = wmma16(pa, *(const h16*)(vbase + (long long)(16 + l16) * S), o1);
    o2 = wmma16(pa, *(const h16*)(vbase + (long long)(32 + l16) * S), o2);
    o3 = wmma16(pa, *(const h16*)(vbase + (long long)(48 + l16) * S), o3);
  }
  sumexp += __shfl_xor(sumexp, 16, 32);  // lanes L and L+16 cover row l16
  __syncthreads();
  if (lane < 16) rowred[l16] = 1.0f / sumexp;
  __syncthreads();
  const int b = bh >> 4, h = bh & 15;
  _Float16* cbase = ctxh + (long long)b * S * D + (long long)h * HD;
#pragma unroll
  for (int r = 0; r < 8; ++r) {
    int s = m0 + r + hf * 8;
    float inv = rowred[r + hf * 8];
    _Float16* crow = cbase + (long long)s * D + l16;
    crow[0]  = (_Float16)(o0[r] * inv);
    crow[16] = (_Float16)(o1[r] * inv);
    crow[32] = (_Float16)(o2[r] * inv);
    crow[48] = (_Float16)(o3[r] * inv);
  }
}

// ---------------- layernorm(a + b) -> f32 (+ optional f16 copy) ----------------
__global__ __launch_bounds__(256) void ln_add_kernel(
    const float* __restrict__ a, const float* __restrict__ b2,
    const float* __restrict__ scale, const float* __restrict__ bias,
    float* __restrict__ outf, _Float16* __restrict__ outh) {
  __shared__ float red[256];
  __shared__ float row[1024];
  const int t = threadIdx.x;
  const long long base = (long long)blockIdx.x * D;
  float s = 0.f, s2 = 0.f;
  for (int d = t; d < D; d += 256) {
    float v = a[base + d] + b2[base + d];
    row[d] = v; s += v; s2 += v * v;
  }
  red[t] = s; __syncthreads();
  for (int o = 128; o > 0; o >>= 1) { if (t < o) red[t] += red[t + o]; __syncthreads(); }
  float mean = red[0] * (1.0f / D); __syncthreads();
  red[t] = s2; __syncthreads();
  for (int o = 128; o > 0; o >>= 1) { if (t < o) red[t] += red[t + o]; __syncthreads(); }
  float var = red[0] * (1.0f / D) - mean * mean;
  float inv = rsqrtf(var + 1e-5f);
  for (int d = t; d < D; d += 256) {
    float v = (row[d] - mean) * inv * scale[d] + bias[d];
    outf[base + d] = v;
    if (outh) outh[base + d] = (_Float16)v;
  }
}

// ---------------- router: wave e computes logit e ----------------
__global__ __launch_bounds__(256) void router_kernel(
    const float* __restrict__ x1, const float* __restrict__ rw,
    float* __restrict__ logits_out, float* __restrict__ topp,
    int* __restrict__ eidx) {
  __shared__ float lsh[E];
  const int tok = blockIdx.x;
  const int wave = threadIdx.x >> 5, lane = threadIdx.x & 31;
  const float* xr = x1 + (long long)tok * D;
  const float* wr = rw + wave * D;
  float acc = 0.f;
  for (int d = lane; d < D; d += 32) acc += xr[d] * wr[d];
  for (int off = 16; off > 0; off >>= 1) acc += __shfl_xor(acc, off, 32);
  if (lane == 0) { lsh[wave] = acc; logits_out[(long long)tok * E + wave] = acc; }
  __syncthreads();
  if (threadIdx.x == 0) {
    float mx = lsh[0]; int mi = 0;
    for (int e = 1; e < E; ++e) if (lsh[e] > mx) { mx = lsh[e]; mi = e; }
    float se = 0.f;
    for (int e = 0; e < E; ++e) se += __expf(lsh[e] - mx);
    topp[tok] = 1.0f / se;       // softmax max prob
    eidx[tok] = mi;
  }
}

__global__ void fill_i32(int* __restrict__ p, int n, int v) {
  int i = blockIdx.x * 256 + threadIdx.x;
  if (i < n) p[i] = v;
}

// sequential capacity scan, one thread per batch
__global__ void scan_kernel(const int* __restrict__ eidx,
                            int* __restrict__ tok2slot,
                            int* __restrict__ slot2tok,
                            float* __restrict__ capped) {
  int b = threadIdx.x;
  if (b >= B) return;
  int cnt[E];
  for (int e = 0; e < E; ++e) cnt[e] = 0;
  for (int s = 0; s < S; ++s) {
    int tok = b * S + s;
    int e = eidx[tok];
    int c = cnt[e];
    if (c < CAP) {
      cnt[e] = c + 1;
      int slot = (b * E + e) * CAP + c;
      tok2slot[tok] = slot;
      slot2tok[slot] = tok;
      capped[tok] = (float)e;      // argmax of capacity-masked mask
    } else {
      tok2slot[tok] = -1;
      capped[tok] = 0.0f;          // argmax of all-zero row
    }
  }
}

__global__ __launch_bounds__(256) void gather_kernel(
    const int* __restrict__ slot2tok, const _Float16* __restrict__ x1h,
    _Float16* __restrict__ xinh) {
  int slot = blockIdx.x;
  int tok = slot2tok[slot];
  _Float16* dst = xinh + (long long)slot * D;
  if (tok >= 0) {
    const _Float16* src = x1h + (long long)tok * D;
    for (int d = threadIdx.x; d < D; d += 256) dst[d] = src[d];
  } else {
    for (int d = threadIdx.x; d < D; d += 256) dst[d] = (_Float16)0.0f;
  }
}

// combine + final layernorm -> d_out
__global__ __launch_bounds__(256) void combine_ln_kernel(
    const float* __restrict__ x1, const float* __restrict__ eo,
    const int* __restrict__ tok2slot, const float* __restrict__ topp,
    const float* __restrict__ scale, const float* __restrict__ bias,
    float* __restrict__ out) {
  __shared__ float red[256];
  __shared__ float row[1024];
  const int tok = blockIdx.x, t = threadIdx.x;
  const long long base = (long long)tok * D;
  const int slot = tok2slot[tok];
  const float tp = topp[tok];
  float s = 0.f, s2 = 0.f;
  for (int d = t; d < D; d += 256) {
    float xs = x1[base + d];
    float v = (slot >= 0) ? eo[(long long)slot * D + d] : xs;
    float val = xs + tp * v;      // residual + top_prob * next_states
    row[d] = val; s += val; s2 += val * val;
  }
  red[t] = s; __syncthreads();
  for (int o = 128; o > 0; o >>= 1) { if (t < o) red[t] += red[t + o]; __syncthreads(); }
  float mean = red[0] * (1.0f / D); __syncthreads();
  red[t] = s2; __syncthreads();
  for (int o = 128; o > 0; o >>= 1) { if (t < o) red[t] += red[t + o]; __syncthreads(); }
  float var = red[0] * (1.0f / D) - mean * mean;
  float inv = rsqrtf(var + 1e-5f);
  for (int d = t; d < D; d += 256)
    out[base + d] = (row[d] - mean) * inv * scale[d] + bias[d];
}

// ---------------- host launch ----------------
extern "C" void kernel_launch(void* const* d_in, const int* in_sizes, int n_in,
                              void* d_out, int out_size, void* d_ws, size_t ws_size,
                              hipStream_t stream) {
  const float* x        = (const float*)d_in[0];
  const float* in_w     = (const float*)d_in[1];
  const float* in_b     = (const float*)d_in[2];
  const float* out_w    = (const float*)d_in[3];
  const float* out_b    = (const float*)d_in[4];
  const float* n1s      = (const float*)d_in[5];
  const float* n1b      = (const float*)d_in[6];
  const float* n2s      = (const float*)d_in[7];
  const float* n2b      = (const float*)d_in[8];
  const float* router_w = (const float*)d_in[9];
  const float* wi       = (const float*)d_in[10];
  const float* wo       = (const float*)d_in[11];

  // d_out layout: out[4,1024,1024] | logits[4,1024,8] | capped_idx[4,1024]
  float* out_main   = (float*)d_out;
  float* out_logits = out_main + (long long)TOK * D;
  float* out_capped = out_logits + (long long)TOK * E;

  // bump allocator over d_ws (256B aligned)
  size_t off = 0;
  auto alloc = [&](size_t bytes) -> void* {
    void* p = (char*)d_ws + off;
    off = (off + bytes + 255) & ~(size_t)255;
    return p;
  };
  _Float16* xh     = (_Float16*)alloc((size_t)TOK * D * 2);
  _Float16* winh   = (_Float16*)alloc((size_t)3 * D * D * 2);
  _Float16* wouth  = (_Float16*)alloc((size_t)D * D * 2);
  _Float16* qh     = (_Float16*)alloc((size_t)B * H * S * HD * 2);
  _Float16* kh     = (_Float16*)alloc((size_t)B * H * S * HD * 2);
  _Float16* vth    = (_Float16*)alloc((size_t)B * H * HD * S * 2);
  _Float16* ctxh   = (_Float16*)alloc((size_t)TOK * D * 2);
  float*    attnout= (float*)   alloc((size_t)TOK * D * 4);
  float*    x1     = (float*)   alloc((size_t)TOK * D * 4);
  _Float16* x1h    = (_Float16*)alloc((size_t)TOK * D * 2);
  float*    topp   = (float*)   alloc((size_t)TOK * 4);
  int*      eidx   = (int*)     alloc((size_t)TOK * 4);
  int*      tok2slot=(int*)     alloc((size_t)TOK * 4);
  int*      slot2tok=(int*)     alloc((size_t)SLOTS * 4);
  _Float16* xinh   = (_Float16*)alloc((size_t)SLOTS * D * 2);
  _Float16* wih    = (_Float16*)alloc((size_t)E * F * D * 2);
  _Float16* woh    = (_Float16*)alloc((size_t)E * D * F * 2);
  _Float16* hh     = (_Float16*)alloc((size_t)SLOTS * F * 2);
  float*    eo     = (float*)   alloc((size_t)SLOTS * D * 4);
  (void)ws_size; (void)in_sizes; (void)n_in; (void)out_size;

  // 1) fp32 -> fp16 conversions (+ transposes for NT layout)
  cvt_f32_to_f16<<<2048, 256, 0, stream>>>(x, xh, (long long)TOK * D);
  cvt_f32_to_f16<<<2048, 256, 0, stream>>>(in_w, winh, (long long)3 * D * D);
  cvt_f32_to_f16<<<1024, 256, 0, stream>>>(out_w, wouth, (long long)D * D);
  trans_wi_kernel<<<E * D, 256, 0, stream>>>(wi, wih);
  trans_wo_kernel<<<E * F, 256, 0, stream>>>(wo, woh);

  // 2) QKV projection -> attention layouts (32x64 tiles)
  qkv_kernel<<<dim3(TOK / 32, 3 * D / 64, 1), 32, 0, stream>>>(
      xh, winh, in_b, qh, kh, vth);

  // 3) attention (wave per 16-query tile, per b*h)
  attn_kernel<<<dim3(S / 16, B * H, 1), 32, 0, stream>>>(qh, kh, vth, ctxh);

  // 4) out projection (f32 out)
  gemm_nt<0, 0><<<dim3(TOK / 32, D / 64, 1), 32, 0, stream>>>(
      ctxh, wouth, out_b, attnout, D, D, 0LL, 0LL, 0LL, 0);

  // 5) LN1(x + attn) -> x1 (f32 + f16)
  ln_add_kernel<<<TOK, 256, 0, stream>>>(x, attnout, n1s, n1b, x1, x1h);

  // 6) router (logits straight into d_out)
  router_kernel<<<TOK, 256, 0, stream>>>(x1, router_w, out_logits, topp, eidx);

  // 7) capacity scan + dispatch tables
  fill_i32<<<(SLOTS + 255) / 256, 256, 0, stream>>>(slot2tok, SLOTS, -1);
  scan_kernel<<<1, 32, 0, stream>>>(eidx, tok2slot, slot2tok, out_capped);

  // 8) gather tokens into expert buffers
  gather_kernel<<<SLOTS, 256, 0, stream>>>(slot2tok, x1h, xinh);

  // 9) expert FFN: GELU(xin @ wi) -> hh (f16); then hh @ wo -> eo (f32)
  gemm_nt<1, 1><<<dim3(CAP / 32, F / 64, B * E), 32, 0, stream>>>(
      xinh, wih, nullptr, hh, F, D,
      (long long)CAP * D, (long long)F * D, (long long)CAP * F, E - 1);
  gemm_nt<0, 0><<<dim3(CAP / 32, D / 64, B * E), 32, 0, stream>>>(
      hh, woh, nullptr, eo, D, F,
      (long long)CAP * F, (long long)D * F, (long long)CAP * D, E - 1);

  // 10) combine + LN2 -> d_out
  combine_ln_kernel<<<TOK, 256, 0, stream>>>(
      x1, eo, tok2slot, topp, n2s, n2b, out_main);
}
